// MathematicallyRigorousConstraintSolver_69011534512613
// MI455X (gfx1250) — matrix operand order repaired and checked by reading, not compiled, not run on hardware
//
#include <hip/hip_runtime.h>

typedef __attribute__((ext_vector_type(16))) _Float16 v16h;
typedef __attribute__((ext_vector_type(8)))  float    v8f;

#define Q_POINTS 1024          // midpoint-rule quadrature points
#define SIG_SCALE 100.0f

// Stable branch-free logistic: rcp(inf) = 0 handles z << 0 exactly.
__device__ __forceinline__ float fast_sigmoid(float z) {
    return __builtin_amdgcn_rcpf(1.0f + __expf(-z));
}

// lgamma for x >= 1 via shift-by-3 Stirling (error < 1e-6 on [1, 45]).
__device__ __forceinline__ float lgamma_pos(float x) {
    const float HALF_LN_2PI = 0.918938533204672742f;
    float z   = x + 3.0f;
    float iz  = __builtin_amdgcn_rcpf(z);
    float iz2 = iz * iz;
    float ser = fmaf(-(1.0f / 360.0f) * iz2, iz, (1.0f / 12.0f) * iz);
    float lg  = fmaf(z - 0.5f, __logf(z), -z) + HALF_LN_2PI + ser;
    return lg - __logf(x * (x + 1.0f) * (x + 2.0f));
}

// One wave32 per 16 output elements.
// prob[m] = (1/Q) * sum_q BetaPDF(x_q; a_m, b_m) * sigmoid(SCALE*(thr - x_q))
// computed as a 16x16x32 f16 WMMA matvec; per-q logs/weights staged in LDS.
__global__ __launch_bounds__(256)
void beta_smoothed_cdf_wmma(const float* __restrict__ alpha,
                            const float* __restrict__ beta,
                            const int*   __restrict__ coord_idx_p,
                            const int*   __restrict__ value_p,
                            const int*   __restrict__ offset_p,
                            float*       __restrict__ out) {
    __shared__ _Float16 s_w[Q_POINTS];      // sigmoid weights (f16, B-fragment source)
    __shared__ float    s_lnx[Q_POINTS];    // ln(x_q)
    __shared__ float    s_ln1mx[Q_POINTS];  // ln(1 - x_q)

    const int   coord = *coord_idx_p;
    const float thr   = fminf(fmaxf((float)(*value_p + *offset_p) * (1.0f / 1000.0f),
                                    0.0f), 1.0f);
    const float invQ  = 1.0f / (float)Q_POINTS;

    // ---- stage per-q tables: 256 threads x 4 points ----
    {
        const int q0 = threadIdx.x * 4;
        #pragma unroll
        for (int j = 0; j < 4; ++j) {
            const int   q = q0 + j;
            const float x = ((float)q + 0.5f) * invQ;
            s_lnx[q]   = __logf(x);
            s_ln1mx[q] = __logf(1.0f - x);
            s_w[q]     = (_Float16)fast_sigmoid((thr - x) * SIG_SCALE);
        }
    }
    __syncthreads();

    const int lane = threadIdx.x & 31;
    const int tile = blockIdx.x * (blockDim.x >> 5) + (threadIdx.x >> 5); // 0..511
    const int row  = lane & 15;   // M index within the A-fragment
    const int half = lane >> 4;   // lane-half selects K block
    const int elem = tile * 16 + row;

    // Per-row Beta(a,b) parameters and log-normalizer (a,b in [1,21])
    const float a   = alpha[elem * 4 + coord];
    const float b   = beta [elem * 4 + coord];
    const float lnB = lgamma_pos(a) + lgamma_pos(b) - lgamma_pos(a + b);
    const float am1 = a - 1.0f;
    const float bm1 = b - 1.0f;

    v8f acc = {};

    for (int t = 0; t < Q_POINTS / 32; ++t) {
        // A K-slot mapping (16-bit 16x32): half0 -> K 0..15 ; half1 -> K 16..31
        const int qa0 = t * 32 + 16 * half;       // slots i = 0..7
        const int qa1 = t * 32 + 16 * half + 8;   // slots i = 8..15
        // B K-slot mapping (32x16): lanes 0-15 hold K 0..15, lanes 16-31 hold 16..31
        const int qb  = t * 32 + 16 * half;       // 16 contiguous halves, 32B aligned

        v16h bfrag = *(const v16h*)&s_w[qb];

        v16h afrag;
        #pragma unroll
        for (int i = 0; i < 16; ++i) {
            const int   q    = (i < 8) ? (qa0 + i) : (qa1 + i - 8);
            const float lpdf = fmaf(am1, s_lnx[q], fmaf(bm1, s_ln1mx[q], -lnB));
            afrag[i] = (_Float16)__expf(lpdf);    // max pdf ~21, safe in f16
        }

        // D = A x B + C, f32 accumulate (EXEC all-1s: uniform control flow)
        acc = __builtin_amdgcn_wmma_f32_16x16x32_f16(
            /*neg_a=*/false, afrag, /*neg_b=*/false, bfrag,
            /*c_mod=*/(short)0, acc, /*reuse_a=*/false, /*reuse_b=*/false);
    }

    // C/D layout: lane 0 holds M=0..7 (VGPR r -> M=r), lane 16 holds M=8..15.
    if (row == 0) {
        #pragma unroll
        for (int r = 0; r < 8; ++r) {
            float p = acc[r] * invQ;
            p = fminf(fmaxf(p, 1e-8f), 1.0f - 1e-8f);
            out[tile * 16 + half * 8 + r] = p;
        }
    }
}

extern "C" void kernel_launch(void* const* d_in, const int* in_sizes, int n_in,
                              void* d_out, int out_size, void* d_ws, size_t ws_size,
                              hipStream_t stream) {
    const float* alpha  = (const float*)d_in[0];
    const float* beta   = (const float*)d_in[1];
    const int*   coord  = (const int*)d_in[2];
    const int*   value  = (const int*)d_in[3];
    const int*   offset = (const int*)d_in[4];
    float*       out    = (float*)d_out;

    // 8192 outputs, 16 per wave, 8 waves per 256-thread block -> 64 blocks
    dim3 grid(64), block(256);
    beta_smoothed_cdf_wmma<<<grid, block, 0, stream>>>(alpha, beta, coord, value, offset, out);
}
